// DecompFEDformerEncoder_36086315221243
// MI455X (gfx1250) — matrix operand order repaired and checked by reading, not compiled, not run on hardware
//
#include <hip/hip_runtime.h>
#include <hip/hip_bf16.h>
#include <math.h>

// ---------------- problem constants ----------------
#define BB    32
#define LSEQ  1024
#define FEAT  128
#define DM    512
#define DFF   2048
#define NH    8
#define EH    64
#define NMODE 64
#define NCLS  128
#define ROWS  (BB*LSEQ)      // 32768
#define FFCHUNK 8192

typedef __bf16 v16bf __attribute__((ext_vector_type(16)));
typedef float  v8f   __attribute__((ext_vector_type(8)));

union FragU { v16bf v; unsigned int u[8]; };

__device__ __forceinline__ unsigned short bfbits(float f) {
    __bf16 h = (__bf16)f;                       // hardware v_cvt, RNE
    return __builtin_bit_cast(unsigned short, h);
}

enum { EPI_NONE = 0, EPI_BIAS = 1, EPI_BIAS_PE = 2, EPI_BIAS_ADD = 3, EPI_GELU = 4, EPI_ADD = 5 };

// ---------------------------------------------------------------------------
// bf16-WMMA GEMM:  C[b] = A[b] (MxK bf16, K-contig) * Bt[b]^T (Bt: NxK bf16)
// 128x128 C-tile, 8 waves, wave tile 64x32, K-step 32.
// Double-buffered LDS; staging via CDNA5 async global->LDS DMA (ASYNCcnt):
// iteration t issues tile t+1's async copies, waits asynccnt<=4 (tile t done,
// tile t+1 in flight), and runs WMMA on tile t while t+1 streams in.
// Requires M%128==0, N%128==0, K%32==0 (true for every call below).
// ---------------------------------------------------------------------------
__global__ __launch_bounds__(256)
void gemm_wmma_bf16(const unsigned short* __restrict__ A, long long sA, int lda,
                    const unsigned short* __restrict__ Bt, long long sB, int ldb,
                    float* __restrict__ C, long long sC, int ldc,
                    int K, int epi, int c_bf16,
                    const float* __restrict__ bias,
                    const float* __restrict__ addend, long long sAdd,
                    const float* __restrict__ pe)
{
    __shared__ unsigned short lA[2][128][40];   // [buf][m][k] bf16, pad-40 stride (80B)
    __shared__ unsigned short lB[2][128][40];   // [buf][n][k] bf16

    const int tid = threadIdx.x;
    const int bz  = blockIdx.z;
    A  += (long long)bz * sA;
    Bt += (long long)bz * sB;
    C  += (long long)bz * sC;
    if (addend) addend += (long long)bz * sAdd;

    const int m0 = blockIdx.y * 128, n0 = blockIdx.x * 128;
    const int wid = tid >> 5, lane = tid & 31;
    const int wrow = wid >> 2, wcol = wid & 3;   // 2x4 wave grid -> 64x32 per wave

    v8f zero = {};
    v8f acc[4][2];
#pragma unroll
    for (int i = 0; i < 4; ++i)
#pragma unroll
        for (int j = 0; j < 2; ++j) acc[i][j] = zero;

    // stage one 128x32 A tile + 128x32 B tile into LDS buffer `buf`:
    // 4 async b128 instructions per wave (2 per thread per matrix)
    auto stage = [&](int buf, int k0) {
#pragma unroll
        for (int i = 0; i < 2; ++i) {
            int idx = tid + i * 256;
            int r = idx >> 1, h = idx & 1;
            unsigned long long ga = (unsigned long long)(A + (size_t)(m0 + r) * lda + k0 + h * 16);
            unsigned la = (unsigned)(unsigned long long)&lA[buf][r][h * 16];  // flat LDS low 32b = LDS offset
            asm volatile("global_load_async_to_lds_b128 %0, %1, off"
                         :: "v"(la), "v"(ga) : "memory");
            unsigned long long gb = (unsigned long long)(Bt + (size_t)(n0 + r) * ldb + k0 + h * 16);
            unsigned lb = (unsigned)(unsigned long long)&lB[buf][r][h * 16];
            asm volatile("global_load_async_to_lds_b128 %0, %1, off"
                         :: "v"(lb), "v"(gb) : "memory");
        }
    };

    const int nk = K >> 5;
    stage(0, 0);

    for (int t = 0; t < nk; ++t) {
        const int cur = t & 1;
        if (t + 1 < nk) {
            stage(cur ^ 1, (t + 1) * 32);
            if (t + 2 < nk) {   // speculative prefetch two tiles ahead (global_prefetch_b8)
                __builtin_prefetch(A  + (size_t)(m0 + (tid >> 1)) * lda + (t + 2) * 32, 0, 1);
                __builtin_prefetch(Bt + (size_t)(n0 + (tid >> 1)) * ldb + (t + 2) * 32, 0, 1);
            }
            // async loads complete in order: wait until only the 4 just-issued remain
            asm volatile("s_wait_asynccnt 0x4" ::: "memory");
        } else {
            asm volatile("s_wait_asynccnt 0x0" ::: "memory");
        }
        __syncthreads();

        // ---- assemble WMMA fragments per documented CDNA5 VGPR layouts ----
        FragU af[4], bf[2];
        const int hl = lane & 15, hh = lane >> 4;
#pragma unroll
        for (int mt = 0; mt < 4; ++mt) {
            int row = wrow * 64 + mt * 16 + hl;                 // M = hl
            const unsigned int* pr = (const unsigned int*)&lA[cur][row][0];
            int kb2 = hh * 4;                                   // K base 0 / 8 (bf16) in dwords
#pragma unroll
            for (int j = 0; j < 4; ++j) {
                af[mt].u[j]     = pr[kb2 + j];                  // K = hh*8 + 2j,2j+1
                af[mt].u[4 + j] = pr[kb2 + 8 + j];              // K = hh*8 + 16 + 2j
            }
        }
#pragma unroll
        for (int nt = 0; nt < 2; ++nt) {
            int col = wcol * 32 + nt * 16 + hl;                 // N = hl
            const unsigned int* pr = (const unsigned int*)&lB[cur][col][0];
            int kb2 = hh * 4;                                   // K = hh*8 + 2j (j<4), hh*8 + 16 + 2(j-4) (j>=4)
#pragma unroll
            for (int j = 0; j < 8; ++j) bf[nt].u[j] = pr[kb2 + j + (j & 4)];
        }

#pragma unroll
        for (int mt = 0; mt < 4; ++mt)
#pragma unroll
            for (int nt = 0; nt < 2; ++nt)
                acc[mt][nt] = __builtin_amdgcn_wmma_f32_16x16x32_bf16(
                    false, af[mt].v, false, bf[nt].v, (short)0, acc[mt][nt], false, false);

        // all waves must finish reading buffer `cur` before iteration t+1
        // stages into it again
        __syncthreads();
    }

    // ---- epilogue: C/D layout: lane n = lane&15, M = (lane>>4)*8 + vgpr ----
    const int nl = lane & 15, mh = lane >> 4;
#pragma unroll
    for (int mt = 0; mt < 4; ++mt) {
#pragma unroll
        for (int nt = 0; nt < 2; ++nt) {
            int gcol  = n0 + wcol * 32 + nt * 16 + nl;
            int grow0 = m0 + wrow * 64 + mt * 16 + mh * 8;
#pragma unroll
            for (int v = 0; v < 8; ++v) {
                int grow = grow0 + v;
                float val = acc[mt][nt][v];
                if      (epi == EPI_BIAS)     val += bias[gcol];
                else if (epi == EPI_BIAS_PE)  val += bias[gcol] + pe[(size_t)(grow & (LSEQ - 1)) * DM + gcol];
                else if (epi == EPI_BIAS_ADD) val += bias[gcol] + addend[(size_t)grow * ldc + gcol];
                else if (epi == EPI_GELU)     val = 0.5f * val * (1.0f + erff(val * 0.70710678118654752f));
                else if (epi == EPI_ADD)      val += addend[(size_t)grow * ldc + gcol];
                if (c_bf16) ((unsigned short*)C)[(size_t)grow * ldc + gcol] = bfbits(val);
                else        C[(size_t)grow * ldc + gcol] = val;
            }
        }
    }
}

// ---------------------------------------------------------------------------
// f32 -> bf16 flat convert (paired stores)
// ---------------------------------------------------------------------------
__global__ void k_cvt(const float* __restrict__ in, unsigned short* __restrict__ out, size_t n)
{
    size_t i = ((size_t)blockIdx.x * blockDim.x + threadIdx.x) * 2;
    if (i >= n) return;
    const float2 v = *(const float2*)(in + i);
    unsigned u = (unsigned)bfbits(v.x) | ((unsigned)bfbits(v.y) << 16);
    *(unsigned*)(out + i) = u;
}

// ---------------------------------------------------------------------------
// tiled transpose + convert: out[c*R + r] = bf16(in[r*Cc + c]); batched over z
// block (32,8); grid (Cc/32, R/32, batch)
// ---------------------------------------------------------------------------
__global__ void k_cvt_t(const float* __restrict__ in, unsigned short* __restrict__ out,
                        int R, int Cc, long long sIn, long long sOut)
{
    __shared__ float tile[32][33];
    int b = blockIdx.z;
    in  += (long long)b * sIn;
    out += (long long)b * sOut;
    int r0 = blockIdx.y * 32, c0 = blockIdx.x * 32;
    int tx = threadIdx.x, ty = threadIdx.y;
#pragma unroll
    for (int i = 0; i < 32; i += 8)
        tile[ty + i][tx] = in[(size_t)(r0 + ty + i) * Cc + c0 + tx];
    __syncthreads();
#pragma unroll
    for (int i = 0; i < 32; i += 8)
        out[(size_t)(c0 + ty + i) * R + r0 + tx] = bfbits(tile[tx][ty + i]);
}

// ---------------------------------------------------------------------------
// Series decomposition: moving average k=25 with replicate padding.
// ---------------------------------------------------------------------------
__global__ void k_decomp(const float* __restrict__ in, float* __restrict__ seas,
                         float* __restrict__ trend, int Lr, int Dd, int accum, int n)
{
    int idx = blockIdx.x * blockDim.x + threadIdx.x;
    if (idx >= n) return;
    int d = idx % Dd;
    int l = (idx / Dd) % Lr;
    int b = idx / (Dd * Lr);
    const float* col = in + (size_t)b * Lr * Dd + d;
    float s = 0.f;
#pragma unroll
    for (int j = -12; j <= 12; ++j) {
        int ll = l + j; ll = ll < 0 ? 0 : (ll >= Lr ? Lr - 1 : ll);
        s += col[(size_t)ll * Dd];
    }
    float m = s * (1.0f / 25.0f);
    seas[idx] = col[(size_t)l * Dd] - m;
    if (accum) trend[idx] += m; else trend[idx] = m;
}

// ---------------------------------------------------------------------------
// pe (f32) + DFT matrix [2m+ri][l] (bf16) + transposed iDFT matrix [l][2m+ri] (bf16)
// dft [2m][l]  =  cos(wml),  [2m+1][l] = -sin(wml)
// idftT[l][2m] =  s_m cos(wml)/L, [l][2m+1] = -s'_m sin(wml)/L ; s_0=1, s'_0=0, else 2
// ---------------------------------------------------------------------------
__global__ void k_init_mats(float* __restrict__ pe, unsigned short* __restrict__ dft,
                            unsigned short* __restrict__ idftT)
{
    const int NPE = LSEQ * DM, NDFT = 2 * NMODE * LSEQ;
    int idx = blockIdx.x * blockDim.x + threadIdx.x;
    const float TWO_PI = 6.28318530717958647692f;
    if (idx < NPE) {
        int l = idx / DM, c = idx % DM;
        int i = c >> 1;
        float div = expf((float)(2 * i) * (-logf(10000.0f) / (float)DM));
        float ang = (float)l * div;
        pe[idx] = (c & 1) ? cosf(ang) : sinf(ang);
    } else if (idx < NPE + NDFT) {
        int t = idx - NPE;
        int j = t / LSEQ, l = t % LSEQ;
        int m = j >> 1;
        int ph = (m * l) & (LSEQ - 1);                 // exact modular arg reduction
        float ang = (float)ph * (TWO_PI / (float)LSEQ);
        dft[t] = bfbits((j & 1) ? -sinf(ang) : cosf(ang));
    } else if (idx < NPE + 2 * NDFT) {
        int t = idx - NPE - NDFT;
        int l = t / 128, j = t % 128;                  // transposed: [l][j]
        int m = j >> 1;
        int ph = (m * l) & (LSEQ - 1);
        float ang = (float)ph * (TWO_PI / (float)LSEQ);
        float s = (m == 0) ? ((j & 1) ? 0.0f : 1.0f) : 2.0f;
        idftT[t] = bfbits(((j & 1) ? -s * sinf(ang) : s * cosf(ang)) * (1.0f / (float)LSEQ));
    }
}

// ---------------------------------------------------------------------------
// Complex mode mixing: out_sel[b,h,o,m] = sum_i x_sel[b,h,i,m] * w[h,i,o,m]
// xsel f32 [b][2m+ri][he] (from DFT GEMM); Sb bf16 [b][he][2m+ri] (iDFT A operand)
// ---------------------------------------------------------------------------
__global__ void k_fourier_mix(const float* __restrict__ xsel,
                              const float* __restrict__ wre,
                              const float* __restrict__ wim,
                              unsigned short* __restrict__ Sb)
{
    int idx = blockIdx.x * blockDim.x + threadIdx.x;   // (b,h,o,m)
    int m = idx & 63, o = (idx >> 6) & 63, h = (idx >> 12) & 7, b = idx >> 15;
    if (b >= BB) return;
    const float* xr = xsel + ((size_t)b * 128 + 2 * m) * DM + h * EH;
    const float* xi = xr + DM;
    size_t wbase = (((size_t)h * EH) * EH + o) * NMODE + m;   // w[h][i][o][m], i-stride EH*NMODE
    const float* pr = wre + wbase;
    const float* pi = wim + wbase;
    float aR = 0.f, aI = 0.f;
#pragma unroll 8
    for (int i = 0; i < EH; ++i) {
        float xre = xr[i], xim = xi[i];
        float w_r = pr[(size_t)i * (EH * NMODE)];
        float w_i = pi[(size_t)i * (EH * NMODE)];
        aR += xre * w_r - xim * w_i;
        aI += xre * w_i + xim * w_r;
    }
    unsigned short* out = Sb + ((size_t)b * DM + h * EH + o) * 128;
    out[2 * m]     = bfbits(aR);
    out[2 * m + 1] = bfbits(aI);
}

// enc[b,d] = mean_l( x[b,l,d] + trend[b,l,d] )
__global__ void k_mean(const float* __restrict__ x, const float* __restrict__ t,
                       float* __restrict__ enc)
{
    int idx = blockIdx.x * blockDim.x + threadIdx.x;
    if (idx >= BB * DM) return;
    int d = idx % DM, b = idx / DM;
    const float* px = x + (size_t)b * LSEQ * DM + d;
    const float* pt = t + (size_t)b * LSEQ * DM + d;
    float s = 0.f;
    for (int l = 0; l < LSEQ; ++l) s += px[(size_t)l * DM] + pt[(size_t)l * DM];
    enc[idx] = s * (1.0f / (float)LSEQ);
}

// out[b,c] = enc[b,:] . cls_w[:,c] + cls_b[c]
__global__ void k_classify(const float* __restrict__ enc, const float* __restrict__ cw,
                           const float* __restrict__ cb, float* __restrict__ out)
{
    int idx = blockIdx.x * blockDim.x + threadIdx.x;
    if (idx >= BB * NCLS) return;
    int c = idx % NCLS, b = idx / NCLS;
    float a = cb[c];
    const float* e = enc + (size_t)b * DM;
    for (int d = 0; d < DM; ++d) a += e[d] * cw[(size_t)d * NCLS + c];
    out[idx] = a;
}

// ---------------------------------------------------------------------------
extern "C" void kernel_launch(void* const* d_in, const int* in_sizes, int n_in,
                              void* d_out, int out_size, void* d_ws, size_t ws_size,
                              hipStream_t stream)
{
    (void)in_sizes; (void)n_in; (void)out_size; (void)ws_size;

    const float* x_enc   = (const float*)d_in[0];
    const float* trend_w = (const float*)d_in[25];
    const float* trend_b = (const float*)d_in[26];
    const float* seas_w  = (const float*)d_in[27];
    const float* seas_b  = (const float*)d_in[28];
    const float* cls_w   = (const float*)d_in[29];
    const float* cls_b   = (const float*)d_in[30];

    // ---- workspace bump allocator ----
    char* wp = (char*)d_ws;
    auto alloc = [&](size_t bytes) -> void* {
        void* p = (void*)wp;
        wp += (bytes + 255) & ~((size_t)255);
        return p;
    };
    // f32 buffers
    float* pe    = (float*)alloc((size_t)LSEQ * DM * 4);
    float* xS    = (float*)alloc((size_t)ROWS * FEAT * 4);
    float* xT    = (float*)alloc((size_t)ROWS * FEAT * 4);
    float* bufX  = (float*)alloc((size_t)ROWS * DM * 4);
    float* bufT  = (float*)alloc((size_t)ROWS * DM * 4);
    float* bufA  = (float*)alloc((size_t)ROWS * DM * 4);
    float* bufB  = (float*)alloc((size_t)ROWS * DM * 4);
    float* xsel  = (float*)alloc((size_t)BB * 128 * DM * 4);
    float* encb  = (float*)alloc((size_t)BB * DM * 4);
    // bf16 buffers
    unsigned short* dftm   = (unsigned short*)alloc((size_t)128 * LSEQ * 2);  // [j][l]
    unsigned short* idftmT = (unsigned short*)alloc((size_t)LSEQ * 128 * 2);  // [l][j]
    unsigned short* xSb    = (unsigned short*)alloc((size_t)ROWS * FEAT * 2);
    unsigned short* xTb    = (unsigned short*)alloc((size_t)ROWS * FEAT * 2);
    unsigned short* bufXb  = (unsigned short*)alloc((size_t)ROWS * DM * 2);
    unsigned short* bfScr  = (unsigned short*)alloc((size_t)ROWS * DM * 2);   // qT / attn-flat
    unsigned short* Sb     = (unsigned short*)alloc((size_t)BB * DM * 128 * 2);
    unsigned short* ybufb  = (unsigned short*)alloc((size_t)FFCHUNK * DFF * 2);
    unsigned short* twT    = (unsigned short*)alloc((size_t)DM * FEAT * 2);
    unsigned short* swT    = (unsigned short*)alloc((size_t)DM * FEAT * 2);
    unsigned short* qwT    = (unsigned short*)alloc((size_t)DM * DM * 2);
    unsigned short* owT    = (unsigned short*)alloc((size_t)DM * DM * 2);
    unsigned short* c1T    = (unsigned short*)alloc((size_t)DFF * DM * 2);
    unsigned short* c2T    = (unsigned short*)alloc((size_t)DM * DFF * 2);

    auto gemm = [&](const unsigned short* A, long long sA, int lda,
                    const unsigned short* Bt, long long sB, int ldb,
                    float* C, long long sC, int ldc,
                    int M, int N, int K, int batch, int epi, int c_bf16,
                    const float* bias, const float* add, long long sAdd, const float* pe_) {
        dim3 g((unsigned)(N / 128), (unsigned)(M / 128), (unsigned)batch);
        gemm_wmma_bf16<<<g, 256, 0, stream>>>(A, sA, lda, Bt, sB, ldb, C, sC, ldc,
                                              K, epi, c_bf16, bias, add, sAdd, pe_);
    };
    auto cvt = [&](const float* in, unsigned short* out, size_t n) {
        k_cvt<<<(unsigned)((n / 2 + 255) / 256), 256, 0, stream>>>(in, out, n);
    };
    auto cvt_t = [&](const float* in, unsigned short* out, int R, int Cc, int batch,
                     long long sIn, long long sOut) {
        dim3 g((unsigned)(Cc / 32), (unsigned)(R / 32), (unsigned)batch);
        k_cvt_t<<<g, dim3(32, 8), 0, stream>>>(in, out, R, Cc, sIn, sOut);
    };

    // constants + weight prep (deterministic, every call)
    {
        int tot = LSEQ * DM + 2 * (2 * NMODE * LSEQ);
        k_init_mats<<<(tot + 255) / 256, 256, 0, stream>>>(pe, dftm, idftmT);
    }
    cvt_t(trend_w, twT, FEAT, DM, 1, 0, 0);
    cvt_t(seas_w,  swT, FEAT, DM, 1, 0, 0);

    // input decomposition + embeddings
    {
        int n = ROWS * FEAT;
        k_decomp<<<(n + 255) / 256, 256, 0, stream>>>(x_enc, xS, xT, LSEQ, FEAT, 0, n);
        cvt(xS, xSb, (size_t)n);
        cvt(xT, xTb, (size_t)n);
    }
    gemm(xTb, 0, FEAT, twT, 0, FEAT, bufT, 0, DM, ROWS, DM, FEAT, 1, EPI_BIAS,    0, trend_b, nullptr, 0, nullptr);
    gemm(xSb, 0, FEAT, swT, 0, FEAT, bufX, 0, DM, ROWS, DM, FEAT, 1, EPI_BIAS_PE, 0, seas_b,  nullptr, 0, pe);

    for (int layer = 0; layer < 2; ++layer) {
        const int base = 1 + layer * 12;
        const float* qw  = (const float*)d_in[base + 0];
        const float* qb  = (const float*)d_in[base + 1];
        const float* ow  = (const float*)d_in[base + 6];
        const float* ob  = (const float*)d_in[base + 7];
        const float* fre = (const float*)d_in[base + 8];
        const float* fim = (const float*)d_in[base + 9];
        const float* c1  = (const float*)d_in[base + 10];
        const float* c2  = (const float*)d_in[base + 11];

        // per-layer weight transposes to [N][K] bf16
        cvt_t(qw, qwT, DM, DM, 1, 0, 0);
        cvt_t(ow, owT, DM, DM, 1, 0, 0);
        cvt_t(c1, c1T, DM, DFF, 1, 0, 0);
        cvt_t(c2, c2T, DFF, DM, 1, 0, 0);

        // q projection
        cvt(bufX, bufXb, (size_t)ROWS * DM);
        gemm(bufXb, 0, DM, qwT, 0, DM, bufA, 0, DM, ROWS, DM, DM, 1, EPI_BIAS, 0, qb, nullptr, 0, nullptr);

        // q^T per batch: [1024,512] -> [512,1024] bf16  (DFT B operand)
        cvt_t(bufA, bfScr, LSEQ, DM, BB, (long long)LSEQ * DM, (long long)DM * LSEQ);

        // DFT (low 64 modes): per batch [128,1024] x [1024,512] -> xsel f32
        gemm(dftm, 0, LSEQ, bfScr, (long long)DM * LSEQ, LSEQ,
             xsel, (long long)128 * DM, DM, 128, DM, LSEQ, BB, EPI_NONE, 0, nullptr, nullptr, 0, nullptr);

        // complex mode mixing -> Sb bf16 [b][he][j]
        {
            int n = BB * NH * EH * NMODE;
            k_fourier_mix<<<(n + 255) / 256, 256, 0, stream>>>(xsel, fre, fim, Sb);
        }

        // iDFT: per batch [512,128] x [128,1024]^T -> bufA f32 as [b][he][l]
        // (contiguous [B,H,E,L] == torch's .view(B,L,-1) reinterpretation for free)
        gemm(Sb, (long long)DM * 128, 128, idftmT, 0, 128,
             bufA, (long long)DM * LSEQ, LSEQ, DM, LSEQ, 128, BB, EPI_NONE, 0, nullptr, nullptr, 0, nullptr);

        // output projection + residual:  bufB = x + attn@ow + ob
        cvt(bufA, bfScr, (size_t)ROWS * DM);
        gemm(bfScr, 0, DM, owT, 0, DM, bufB, 0, DM, ROWS, DM, DM, 1, EPI_BIAS_ADD, 0, ob, bufX, 0, nullptr);

        // decomp: x = seasonal(bufB), trend += mean
        {
            int n = ROWS * DM;
            k_decomp<<<(n + 255) / 256, 256, 0, stream>>>(bufB, bufX, bufT, LSEQ, DM, 1, n);
        }

        // FFN in 4 row chunks: y = gelu(x@c1) (bf16); bufB = x + y@c2
        cvt(bufX, bufXb, (size_t)ROWS * DM);
        for (int c = 0; c < ROWS / FFCHUNK; ++c) {
            const unsigned short* Ain = bufXb + (size_t)c * FFCHUNK * DM;
            const float* Aresid = bufX + (size_t)c * FFCHUNK * DM;
            gemm(Ain, 0, DM, c1T, 0, DM, (float*)ybufb, 0, DFF,
                 FFCHUNK, DFF, DM, 1, EPI_GELU, 1, nullptr, nullptr, 0, nullptr);
            gemm(ybufb, 0, DFF, c2T, 0, DFF, bufB + (size_t)c * FFCHUNK * DM, 0, DM,
                 FFCHUNK, DM, DFF, 1, EPI_ADD, 0, nullptr, Aresid, 0, nullptr);
        }

        // decomp again
        {
            int n = ROWS * DM;
            k_decomp<<<(n + 255) / 256, 256, 0, stream>>>(bufB, bufX, bufT, LSEQ, DM, 1, n);
        }
    }

    // enc = mean_l(x + trend); out = enc @ cls_w + cls_b
    k_mean<<<(BB * DM + 255) / 256, 256, 0, stream>>>(bufX, bufT, encb);
    k_classify<<<(BB * NCLS + 255) / 256, 256, 0, stream>>>(encb, cls_w, cls_b, (float*)d_out);
}
